// TextEncoder_41927470743830
// MI455X (gfx1250) — matrix operand order repaired and checked by reading, not compiled, not run on hardware
//
#include <hip/hip_runtime.h>
#include <cstdint>

#define B_ 32
#define S_ 512
#define H_ 512
#define F_ 2048
#define L_ 6
#define NH_ 8
#define HD_ 64
#define OUT_ 80
#define FD_ 256

typedef __attribute__((ext_vector_type(16))) __bf16 v16bf;
typedef __attribute__((ext_vector_type(8)))  float v8f;

__device__ __forceinline__ unsigned short f2bf(float f) {
  unsigned u = __float_as_uint(f);
  u += 0x7FFFu + ((u >> 16) & 1u);
  return (unsigned short)(u >> 16);
}

// Flags
#define GF_RELU   1
#define GF_TSTORE 2   // store C[m][n] -> Out[((m/S)*NoutT + n)*S + (m%S)]
#define GF_ABF16  4   // A operand is bf16 (ushort) instead of f32
#define GF_CBF16  8   // C written as bf16 (ushort*)

// LDS tile row stride: 48 halves = 96B = multiple of 16B so ds_load_b128 frag
// reads are always 16B-aligned.
#define KPAD 48

// Generic tiled WMMA GEMM:  C[z][m][n] = sum_{tap,k} A[z][m+shift(tap)][k] * Wb[z][tap][n][k] (+bias)
// All B operands are pre-packed bf16 with contiguous k (b_ns = row stride, b_ts = tap stride).
// A rows are S_-periodic (conv 'SAME' taps zero-padded at sequence boundaries).
// Double-buffered LDS + register-staged pipeline: next tile's global loads are issued
// before the current tile's 8 WMMAs so HBM/L2 latency hides under the matrix pipe.
__global__ __launch_bounds__(256)
void gemm_wmma_kernel(const float* __restrict__ Af,
                      const unsigned short* __restrict__ Abf,
                      const unsigned short* __restrict__ Wb,
                      const float* __restrict__ bias,
                      float* __restrict__ C,
                      int M, int N, int K, int lda,
                      long long a_z1, long long a_z2,
                      int b_ns, long long b_ts,
                      long long b_z1, long long b_z2,
                      int ldc, long long c_z1, long long c_z2,
                      int zdiv, int ntaps, int flags, int NoutT)
{
  __shared__ unsigned short As[2][128][KPAD];
  __shared__ unsigned short Bs[2][128][KPAD];

  const int tid  = threadIdx.x;
  const int lane = tid & 31;
  const int wv   = tid >> 5;
  const int l16  = lane & 15;
  const int hi   = lane >> 4;            // 0 or 1 (half-wave)
  const int m0   = blockIdx.x * 128;
  const int n0   = blockIdx.y * 128;
  const int z    = blockIdx.z;

  const long long za = (long long)(z / zdiv) * a_z1 + (long long)(z % zdiv) * a_z2;
  const long long zb = (long long)(z / zdiv) * b_z1 + (long long)(z % zdiv) * b_z2;
  const long long zc = (long long)(z / zdiv) * c_z1 + (long long)(z % zdiv) * c_z2;

  const bool abf = (flags & GF_ABF16) != 0;
  const int nk32  = K >> 5;              // K is always a multiple of 32 here
  const int niter = ntaps * nk32;

  v8f acc[8] = {};
  float4  raf[4];                        // staged A (f32 path)
  ushort4 rab[4];                        // staged A (bf16 path)
  ushort4 rbv[4];                        // staged B (always bf16)

  // issue global loads for tile `it` into registers (no LDS yet)
  auto loadTile = [&](int it) {
    const int tap = it / nk32;
    const int k0  = (it - tap * nk32) << 5;
    const int shift = tap - (ntaps >> 1);
    const long long btap = zb + (long long)tap * b_ts;
    #pragma unroll
    for (int j = 0; j < 4; ++j) {
      const int i  = tid + j * 256;      // 1024 slots of 4 elements
      const int r  = i >> 3;
      const int k4 = (i & 7) << 2;
      // ---- A
      const int m = m0 + r;
      raf[j] = make_float4(0.f, 0.f, 0.f, 0.f);
      rab[j] = make_ushort4(0, 0, 0, 0);
      if (m < M) {
        const int t = m & (S_ - 1);
        if ((unsigned)(t + shift) < (unsigned)S_) {
          const long long src = za + (long long)(m + shift) * lda + (k0 + k4);
          if (abf) {
            rab[j] = *(const ushort4*)(Abf + src);
            __builtin_prefetch(Abf + src + 32, 0, 3);
          } else {
            raf[j] = *(const float4*)(Af + src);
            __builtin_prefetch(Af + src + 32, 0, 3);
          }
        }
      }
      // ---- B
      const int n = n0 + r;
      rbv[j] = make_ushort4(0, 0, 0, 0);
      if (n < N) {
        const long long src = btap + (long long)n * b_ns + (k0 + k4);
        rbv[j] = *(const ushort4*)(Wb + src);
        __builtin_prefetch(Wb + src + 32, 0, 3);
      }
    }
  };
  // convert (if needed) + write staged registers to LDS buffer `buf`
  auto storeTile = [&](int buf) {
    #pragma unroll
    for (int j = 0; j < 4; ++j) {
      const int i  = tid + j * 256;
      const int r  = i >> 3;
      const int k4 = (i & 7) << 2;
      ushort4 av = abf ? rab[j]
                       : make_ushort4(f2bf(raf[j].x), f2bf(raf[j].y),
                                      f2bf(raf[j].z), f2bf(raf[j].w));
      *(ushort4*)&As[buf][r][k4] = av;
      *(ushort4*)&Bs[buf][r][k4] = rbv[j];
    }
  };

  loadTile(0);
  storeTile(0);

  for (int it = 0; it < niter; ++it) {
    __syncthreads();                     // tile `it` staged for all waves
    const int buf = it & 1;
    if (it + 1 < niter) loadTile(it + 1);   // global loads in flight during WMMAs

    // ---- A fragment for this wave's 16-row strip (contiguous k = 16*hi + s, same order as B)
    union { v16bf v; unsigned u[8]; } fa;
    const int arow = wv * 16 + l16;
    #pragma unroll
    for (int p = 0; p < 8; ++p) {
      int kA = (p << 1) + (hi ? 16 : 0);
      fa.u[p] = *(const unsigned*)&As[buf][arow][kA];
    }
    // ---- 8 N-subtiles per wave, B-fragment ping-pong (load nt+1 ahead of wmma nt)
    union FB { v16bf v; unsigned u[8]; } fb[2];
    const int kb = hi ? 16 : 0;
    #pragma unroll
    for (int p = 0; p < 8; ++p)
      fb[0].u[p] = *(const unsigned*)&Bs[buf][l16][kb + (p << 1)];
    #pragma unroll
    for (int nt = 0; nt < 8; ++nt) {
      if (nt + 1 < 8) {
        const int bcol = (nt + 1) * 16 + l16;
        #pragma unroll
        for (int p = 0; p < 8; ++p)
          fb[(nt + 1) & 1].u[p] = *(const unsigned*)&Bs[buf][bcol][kb + (p << 1)];
      }
      acc[nt] = __builtin_amdgcn_wmma_f32_16x16x32_bf16(
          false, fa.v, false, fb[nt & 1].v, (short)0, acc[nt], false, false);
    }

    if (it + 1 < niter) storeTile((it + 1) & 1);  // s_wait_loadcnt lands here
  }

  // ---- epilogue: C layout = lane holds column n, vgpr r holds row r + 8*hi
  #pragma unroll
  for (int nt = 0; nt < 8; ++nt) {
    int n = n0 + nt * 16 + l16;
    if (n >= N) continue;
    float bb = bias ? bias[n] : 0.f;
    #pragma unroll
    for (int r = 0; r < 8; ++r) {
      int m = m0 + wv * 16 + r + (hi ? 8 : 0);
      if (m >= M) continue;
      float vv = acc[nt][r] + bb;
      if (flags & GF_RELU) vv = fmaxf(vv, 0.f);
      long long oidx;
      if (flags & GF_TSTORE) {
        int bidx = m >> 9, t = m & (S_ - 1);
        oidx = zc + ((long long)bidx * NoutT + n) * S_ + t;
      } else {
        oidx = zc + (long long)m * ldc + n;
      }
      if (flags & GF_CBF16) ((unsigned short*)C)[oidx] = f2bf(vv);
      else                  C[oidx] = vv;
    }
  }
}

// Repack weights f32 (N,K,ntaps) -> bf16 [tap][n][k] contiguous (ntaps=1: plain convert)
__global__ void repack_w_kernel(const float* __restrict__ in, unsigned short* __restrict__ out,
                                int N, int K, int ntaps)
{
  int flat = blockIdx.x * blockDim.x + threadIdx.x;
  if (flat >= N * K * ntaps) return;
  int k = flat % K;
  int rest = flat / K;
  int n = rest % N;
  int tap = rest / N;
  out[flat] = f2bf(in[((long long)n * K + k) * ntaps + tap]);
}

// softmax over 512-wide score rows; replicates reference mask quirk:
// keep score/8 where k_pos >= length[b], else -1e9. Emits bf16 P.
__global__ __launch_bounds__(256)
void softmax_mask_kernel(const float* __restrict__ Sc, unsigned short* __restrict__ P,
                         const int* __restrict__ lens, int b_base, int rows)
{
  int lane = threadIdx.x & 31;
  int wv   = threadIdx.x >> 5;
  int row  = blockIdx.x * 8 + wv;
  if (row >= rows) return;
  int qz = row / S_;                 // b_local*NH + h
  int b  = b_base + qz / NH_;
  int len = lens[b];
  const float* sr = Sc + (long long)row * S_;
  float vals[16], mx = -3.4e38f;
  #pragma unroll
  for (int i = 0; i < 16; ++i) {
    int j = lane + (i << 5);
    float s = sr[j];
    float v = (j >= len) ? s * 0.125f : -1e9f;   // 1/sqrt(HD)=0.125
    vals[i] = v; mx = fmaxf(mx, v);
  }
  #pragma unroll
  for (int o = 16; o > 0; o >>= 1) mx = fmaxf(mx, __shfl_xor(mx, o, 32));
  float sum = 0.f;
  #pragma unroll
  for (int i = 0; i < 16; ++i) { vals[i] = expf(vals[i] - mx); sum += vals[i]; }
  #pragma unroll
  for (int o = 16; o > 0; o >>= 1) sum += __shfl_xor(sum, o, 32);
  float inv = 1.f / sum;
  unsigned short* pr = P + (long long)row * S_;
  #pragma unroll
  for (int i = 0; i < 16; ++i) pr[lane + (i << 5)] = f2bf(vals[i] * inv);
}

// RoPE in-place on (B,S,H), per-head d=64 rotation; optional bf16 mirror output
__global__ void rope_kernel(float* __restrict__ X, unsigned short* __restrict__ Xbf, int total)
{
  int idx = blockIdx.x * blockDim.x + threadIdx.x;
  if (idx >= total) return;
  const int half = H_ / 2;
  int row = idx / half;
  int c2  = idx - row * half;
  int head = c2 >> 5;
  int j    = c2 & 31;
  int t    = row % S_;
  float theta = __expf(-(float)j * (9.210340371976184f / 32.f)); // 10000^(-2j/64)
  float ang = (float)t * theta;
  float c = cosf(ang), s = sinf(ang);
  long long off = (long long)row * H_ + head * 64 + j;
  float lo = X[off], hv = X[off + 32];
  float nlo = lo * c - hv * s;
  float nhi = hv * c + lo * s;
  X[off]      = nlo;
  X[off + 32] = nhi;
  if (Xbf) { Xbf[off] = f2bf(nlo); Xbf[off + 32] = f2bf(nhi); }
}

// out = LayerNorm(a (+res)) * g + b over last dim C; one wave per row
__global__ __launch_bounds__(256)
void ln_kernel(float* __restrict__ out, const float* __restrict__ a, const float* __restrict__ res,
               const float* __restrict__ g, const float* __restrict__ be, int C, int rows)
{
  int lane = threadIdx.x & 31;
  int wv   = threadIdx.x >> 5;
  int row  = blockIdx.x * 8 + wv;
  if (row >= rows) return;
  int per = C >> 5;                    // 16 (H) or 8 (FD)
  float vals[16];
  const float* ar = a + (long long)row * C;
  const float* rr = res ? res + (long long)row * C : nullptr;
  float sum = 0.f;
  for (int i = 0; i < per; ++i) {
    int j = lane + (i << 5);
    float v = ar[j] + (rr ? rr[j] : 0.f);
    vals[i] = v; sum += v;
  }
  for (int o = 16; o > 0; o >>= 1) sum += __shfl_xor(sum, o, 32);
  float mean = sum / (float)C;
  float vs = 0.f;
  for (int i = 0; i < per; ++i) { float d = vals[i] - mean; vs += d * d; }
  for (int o = 16; o > 0; o >>= 1) vs += __shfl_xor(vs, o, 32);
  float inv = rsqrtf(vs / (float)C + 1e-5f);
  float* orow = out + (long long)row * C;
  for (int i = 0; i < per; ++i) {
    int j = lane + (i << 5);
    orow[j] = (vals[i] - mean) * inv * g[j] + be[j];
  }
}

// out = in * m, where m = (t >= len[b]) ? 1 : 0  (reference quirk replicated)
__global__ void maskcopy_kernel(float* __restrict__ out, const float* __restrict__ in,
                                const int* __restrict__ lens, int C, long long total)
{
  long long idx = (long long)blockIdx.x * blockDim.x + threadIdx.x;
  if (idx >= total) return;
  long long row = idx / C;
  int b = (int)(row / S_), t = (int)(row % S_);
  out[idx] = (t >= lens[b]) ? in[idx] : 0.f;
}

__global__ void embed_kernel(const int* __restrict__ x, const float* __restrict__ emb,
                             float* __restrict__ out)
{
  int idx = blockIdx.x * blockDim.x + threadIdx.x;
  if (idx >= B_ * S_ * H_) return;
  int row = idx / H_, c = idx - row * H_;
  out[idx] = emb[(long long)x[row] * H_ + c] * 22.627416997969522f; // sqrt(512)
}

// duration head: v = (x . w + b) * m ; out = ceil(exp(v))
__global__ void dp_proj_kernel(const float* __restrict__ Xin, const float* __restrict__ w,
                               const float* __restrict__ bsc, const int* __restrict__ lens,
                               float* __restrict__ out)
{
  int idx = blockIdx.x * blockDim.x + threadIdx.x;
  if (idx >= B_ * S_) return;
  int b = idx / S_, t = idx - b * S_;
  const float* xr = Xin + (long long)idx * FD_;
  float acc = bsc[0];
  for (int c = 0; c < FD_; ++c) acc += xr[c] * w[c];
  if (t < lens[b]) acc = 0.f;   // m = (t >= len)
  out[idx] = ceilf(expf(acc));
}

extern "C" void kernel_launch(void* const* d_in, const int* in_sizes, int n_in,
                              void* d_out, int out_size, void* d_ws, size_t ws_size,
                              hipStream_t stream)
{
  (void)in_sizes; (void)n_in; (void)out_size; (void)ws_size;
  const int*   x      = (const int*)d_in[0];
  const int*   lens   = (const int*)d_in[1];
  const float* emb    = (const float*)d_in[2];
  const float* Wq     = (const float*)d_in[3];
  const float* bq     = (const float*)d_in[4];
  const float* Wk     = (const float*)d_in[5];
  const float* bk     = (const float*)d_in[6];
  const float* Wv     = (const float*)d_in[7];
  const float* bv     = (const float*)d_in[8];
  const float* Wo     = (const float*)d_in[9];
  const float* bo     = (const float*)d_in[10];
  const float* c1w    = (const float*)d_in[11];
  const float* c1b    = (const float*)d_in[12];
  const float* c2w    = (const float*)d_in[13];
  const float* c2b    = (const float*)d_in[14];
  const float* ln1g   = (const float*)d_in[15];
  const float* ln1b   = (const float*)d_in[16];
  const float* ln2g   = (const float*)d_in[17];
  const float* ln2b   = (const float*)d_in[18];
  const float* projw  = (const float*)d_in[19];
  const float* projb  = (const float*)d_in[20];
  const float* dpc1w  = (const float*)d_in[21];
  const float* dpc1b  = (const float*)d_in[22];
  const float* dpl1g  = (const float*)d_in[23];
  const float* dpl1b  = (const float*)d_in[24];
  const float* dpc2w  = (const float*)d_in[25];
  const float* dpc2b  = (const float*)d_in[26];
  const float* dpl2g  = (const float*)d_in[27];
  const float* dpl2b  = (const float*)d_in[28];
  const float* dppw   = (const float*)d_in[29];
  const float* dppb   = (const float*)d_in[30];

  float* ws = (float*)d_ws;
  const long long BSH = (long long)B_ * S_ * H_;        // 8388608
  float* h   = ws;                 // residual stream (B,S,H)
  float* a   = ws + BSH;           // attn-out / conv2-out / dp-input
  float* q   = ws + 2 * BSH;
  float* k   = ws + 3 * BSH;
  unsigned short* vt  = (unsigned short*)(ws + 4 * BSH);  // V transposed bf16 (B,H,S)
  unsigned short* kbf = vt + BSH;                          // bf16 mirror of K (B,S,H)
  float* big = ws + 5 * BSH;       // conv mid (B,S,F) / attention scores+P chunk
  unsigned short* wts = (unsigned short*)(ws + 5 * BSH + (long long)B_ * S_ * F_); // repacked weights

  const int BS = B_ * S_;
  const int EB = 256;

  auto repack = [&](const float* w, int N, int K, int ntaps) {
    int tot = N * K * ntaps;
    repack_w_kernel<<<(tot + EB - 1) / EB, EB, 0, stream>>>(w, wts, N, K, ntaps);
  };
  auto gemm = [&](const float* Af, const unsigned short* Abf, const unsigned short* Wb,
                  const float* bias, float* C,
                  int M, int N, int K, int lda, long long az1, long long az2,
                  int bns, long long bts, long long bz1, long long bz2,
                  int ldc, long long cz1, long long cz2,
                  int zdiv, int Z, int ntaps, int flags, int NoutT) {
    dim3 grid((M + 127) / 128, (N + 127) / 128, Z);
    gemm_wmma_kernel<<<grid, 256, 0, stream>>>(Af, Abf, Wb, bias, C, M, N, K, lda,
        az1, az2, bns, bts, bz1, bz2, ldc, cz1, cz2, zdiv, ntaps, flags, NoutT);
  };

  embed_kernel<<<(B_ * S_ * H_ + EB - 1) / EB, EB, 0, stream>>>(x, emb, h);

  for (int l = 0; l < L_; ++l) {
    // Q/K projections: out = h @ W^T + b
    repack(Wq + (long long)l * H_ * H_, H_, H_, 1);
    gemm(h, nullptr, wts, bq + l * H_, q, BS, H_, H_, H_, 0, 0, H_, 0, 0, 0, H_, 0, 0, 1, 1, 1, 0, 0);
    repack(Wk + (long long)l * H_ * H_, H_, H_, 1);
    gemm(h, nullptr, wts, bk + l * H_, k, BS, H_, H_, H_, 0, 0, H_, 0, 0, 0, H_, 0, 0, 1, 1, 1, 0, 0);
    // V projection, stored transposed (B,H,S) as bf16 directly
    repack(Wv + (long long)l * H_ * H_, H_, H_, 1);
    gemm(h, nullptr, wts, bv + l * H_, (float*)vt, BS, H_, H_, H_, 0, 0, H_, 0, 0, 0,
         0, 0, 0, 1, 1, 1, GF_TSTORE | GF_CBF16, H_);

    const int npair = B_ * S_ * (H_ / 2);
    rope_kernel<<<(npair + EB - 1) / EB, EB, 0, stream>>>(q, nullptr, npair);
    rope_kernel<<<(npair + EB - 1) / EB, EB, 0, stream>>>(k, kbf, npair);

    // attention in chunks of 8 batches: scores chunk (67MB) stays L2-resident
    const long long SS = (long long)S_ * S_;
    unsigned short* P = (unsigned short*)(big + (long long)8 * NH_ * SS);
    for (int bc = 0; bc < 4; ++bc) {
      long long boff = (long long)bc * 8 * S_ * H_;
      // scores[z][qs][ks] = q_head @ k_head^T   (z = b_local*NH + h, Z = 64)
      gemm(q + boff, nullptr, kbf + boff, nullptr, big,
           S_, S_, HD_, H_, (long long)S_ * H_, HD_,
           H_, 0, (long long)S_ * H_, HD_,
           S_, (long long)NH_ * SS, SS, NH_, 64, 1, 0, 0);
      int rows = 8 * NH_ * S_;
      softmax_mask_kernel<<<rows / 8, 256, 0, stream>>>(big, P, lens, bc * 8, rows);
      // o_head = P @ V_head : B = Vt rows (d-major), contiguous k
      gemm(nullptr, P, vt + boff, nullptr, a + boff,
           S_, HD_, S_, S_, (long long)NH_ * SS, SS,
           S_, 0, (long long)H_ * S_, (long long)HD_ * S_,
           H_, (long long)S_ * H_, HD_, NH_, 64, 1, GF_ABF16, 0);
    }

    // output projection, residual + LN1
    repack(Wo + (long long)l * H_ * H_, H_, H_, 1);
    gemm(a, nullptr, wts, bo + l * H_, q, BS, H_, H_, H_, 0, 0, H_, 0, 0, 0, H_, 0, 0, 1, 1, 1, 0, 0);
    ln_kernel<<<BS / 8, 256, 0, stream>>>(h, h, q, ln1g + l * H_, ln1b + l * H_, H_, BS);

    // conv1 (H->F, K=3, relu) and conv2 (F->H, K=3) as 3-tap implicit-im2col WMMA GEMMs
    repack(c1w + (long long)l * F_ * H_ * 3, F_, H_, 3);
    gemm(h, nullptr, wts, c1b + l * F_, big,
         BS, F_, H_, H_, 0, 0, H_, (long long)F_ * H_, 0, 0, F_, 0, 0, 1, 1, 3, GF_RELU, 0);
    repack(c2w + (long long)l * H_ * F_ * 3, H_, F_, 3);
    gemm(big, nullptr, wts, c2b + l * H_, a,
         BS, H_, F_, F_, 0, 0, F_, (long long)H_ * F_, 0, 0, H_, 0, 0, 1, 1, 3, 0, 0);
    ln_kernel<<<BS / 8, 256, 0, stream>>>(h, h, a, ln2g + l * H_, ln2b + l * H_, H_, BS);
  }

  // mu = 1x1 conv H->80, stored transposed as (B, OUT, S) into d_out
  repack(projw, OUT_, H_, 1);
  gemm(h, nullptr, wts, projb, (float*)d_out,
       BS, OUT_, H_, H_, 0, 0, H_, 0, 0, 0, 0, 0, 0, 1, 1, 1, GF_TSTORE, OUT_);

  // duration predictor
  long long totH = (long long)BS * H_, totF = (long long)BS * FD_;
  maskcopy_kernel<<<(unsigned)((totH + EB - 1) / EB), EB, 0, stream>>>(a, h, lens, H_, totH);
  repack(dpc1w, FD_, H_, 3);
  gemm(a, nullptr, wts, dpc1b, q, BS, FD_, H_, H_, 0, 0, H_, (long long)FD_ * H_, 0, 0,
       FD_, 0, 0, 1, 1, 3, GF_RELU, 0);
  maskcopy_kernel<<<(unsigned)((totF + EB - 1) / EB), EB, 0, stream>>>(q, q, lens, FD_, totF);
  ln_kernel<<<BS / 8, 256, 0, stream>>>(q, q, nullptr, dpl1g, dpl1b, FD_, BS);
  repack(dpc2w, FD_, FD_, 3);
  gemm(q, nullptr, wts, dpc2b, k, BS, FD_, FD_, FD_, 0, 0, FD_, (long long)FD_ * FD_, 0, 0,
       FD_, 0, 0, 1, 1, 3, GF_RELU, 0);
  maskcopy_kernel<<<(unsigned)((totF + EB - 1) / EB), EB, 0, stream>>>(k, k, lens, FD_, totF);
  ln_kernel<<<BS / 8, 256, 0, stream>>>(k, k, nullptr, dpl2g, dpl2b, FD_, BS);
  dp_proj_kernel<<<(B_ * S_ + EB - 1) / EB, EB, 0, stream>>>(
      k, dppw, dppb, lens, (float*)d_out + (long long)B_ * OUT_ * S_);
}
